// GlobalAttention_57294863729023
// MI455X (gfx1250) — matrix operand order repaired and checked by reading, not compiled, not run on hardware
//
#include <hip/hip_runtime.h>
#include <hip/hip_bf16.h>

// ---------------------------------------------------------------------------
// GlobalAttention collapsed pipeline for MI455X (gfx1250, wave32, WMMA).
//
//  y = cat(graph_emb[idx], edge_emb) @ Wc^T + bc        (Wc = W4 W3 W2 W1)
//  out = segment_softmax(y, idx)
//
//  K1..K3: tiny on-device GEMMs to build Wc (128x384) + bc (128)
//  K4:     G[g,f] = Wc[:,0:256] . graph_emb[g] + bc     (1024x128, L2-resident)
//  K4b:    pre-swizzle Wce = Wc[:,256:384] into per-lane bf16 hi/lo WMMA
//          B-fragments (64 KB) so the main kernel does no weight conversion.
//  K5:     main GEMM (M=500000, N=128, K=128): split-bf16 WMMA
//          (3x v_wmma_f32_16x16x32_bf16 per 32-wide k-chunk ~ fp32 precision),
//          acc initialized from gathered G row, inline segment-max atomics.
//  K6/K7:  exp + segment-sum (atomicAdd), then normalize. d_out holds y/e/out.
// ---------------------------------------------------------------------------

#define N_EDGES   500000
#define N_GRAPHS  1024
#define MB        64          // edges per block in main GEMM
#define PK        136         // padded LDS row stride (bf16 elements)

typedef __attribute__((ext_vector_type(16))) __bf16 v16bf;
typedef __attribute__((ext_vector_type(8)))  __bf16 v8bf;
typedef __attribute__((ext_vector_type(4)))  __bf16 v4bf;
typedef __attribute__((ext_vector_type(8)))  float  v8f;

// monotonic uint encoding of f32 for integer atomicMax
__device__ __forceinline__ unsigned encf(float x) {
  unsigned u = __float_as_uint(x);
  return (u & 0x80000000u) ? ~u : (u | 0x80000000u);
}
__device__ __forceinline__ float decf(unsigned u) {
  u = (u & 0x80000000u) ? (u & 0x7fffffffu) : ~u;
  return __uint_as_float(u);
}

// ------------------------- init scratch (gmax, denom) ----------------------
__global__ void ga_init_kernel(unsigned* __restrict__ gmax, float* __restrict__ denom) {
  int t = blockIdx.x * blockDim.x + threadIdx.x;   // 131072 threads
  gmax[t]  = 0u;      // encodes "most negative"
  denom[t] = 0.0f;
}

// ------------------------- tiny weight-collapse GEMM -----------------------
// C[i,j] = sum_k A[i*K+k] * B[k*N+j]   (row-major, thread per output)
__global__ void ga_gemm_small(const float* __restrict__ A, const float* __restrict__ B,
                              float* __restrict__ C, int M, int N, int K) {
  int id = blockIdx.x * blockDim.x + threadIdx.x;
  if (id >= M * N) return;
  int i = id / N, j = id % N;
  float s = 0.f;
  for (int k = 0; k < K; ++k) s += A[i * K + k] * B[k * N + j];
  C[id] = s;
}

// bc[i] = T2[i,:].b_in + T1[i,:].in_proj_b[512:] + W4[i,:].out_proj_b + b_out[i]
__global__ void ga_bias_kernel(const float* __restrict__ T1, const float* __restrict__ T2,
                               const float* __restrict__ W4, const float* __restrict__ b_in,
                               const float* __restrict__ in_proj_b,
                               const float* __restrict__ out_proj_b,
                               const float* __restrict__ b_out, float* __restrict__ bc) {
  int i = threadIdx.x;   // 128
  float s = b_out[i];
  for (int k = 0; k < 256; ++k)
    s += T2[i * 256 + k] * b_in[k]
       + T1[i * 256 + k] * in_proj_b[512 + k]
       + W4[i * 256 + k] * out_proj_b[k];
  bc[i] = s;
}

// ------------------------- per-graph projection ----------------------------
// G[g,f] = bc[f] + sum_{k<256} Wc[f,k] * graph_emb[g,k]
__global__ void ga_graph_proj(const float* __restrict__ Wc, const float* __restrict__ bc,
                              const float* __restrict__ gemb, float* __restrict__ G) {
  int t = blockIdx.x * blockDim.x + threadIdx.x;   // 131072
  int g = t >> 7, f = t & 127;
  const float* wr = Wc + (long)f * 384;
  const float* xr = gemb + (long)g * 256;
  float s = bc[f];
  for (int k = 0; k < 256; k += 4) {
    float4 w = *(const float4*)(wr + k);
    float4 x = *(const float4*)(xr + k);
    s += w.x * x.x + w.y * x.y + w.z * x.z + w.w * x.w;
  }
  G[t] = s;
}

// ------------------------- pre-swizzled B fragments ------------------------
// Fragment (w, kc, lane) holds Wce[n = w*16 + lane%16][K = kc*32 + (lane/16)*16
// .. +15] as 16 bf16 (hi and lo of the f32 split). Stored contiguously so the
// main kernel's lane loads its v16bf with coalesced 32B global loads.
__global__ void ga_bsplit(const float* __restrict__ Wc, __bf16* __restrict__ Bhw,
                          __bf16* __restrict__ Blw) {
  int item = blockIdx.x * blockDim.x + threadIdx.x;   // 1024 items
  int lane = item & 31;
  int kc   = (item >> 5) & 3;
  int w    = item >> 7;
  int n    = w * 16 + (lane & 15);
  int kb   = kc * 32 + (lane >> 4) * 16;
  const float* src = Wc + (long)n * 384 + 256 + kb;
  __bf16* dh = Bhw + (long)item * 16;
  __bf16* dl = Blw + (long)item * 16;
  #pragma unroll
  for (int q = 0; q < 16; ++q) {
    float x  = src[q];
    __bf16 h = (__bf16)x;
    dh[q] = h;
    dl[q] = (__bf16)(x - (float)h);
  }
}

// ------------------------- main GEMM: split-bf16 WMMA ----------------------
// y[e,n] = G[idx[e],n] + sum_{k<128} Wce[n,k]*edge[e,k]
__global__ __launch_bounds__(256) void ga_main_gemm(
    const float* __restrict__ edge, const int* __restrict__ gidx,
    const __bf16* __restrict__ Bhw, const __bf16* __restrict__ Blw,
    const float* __restrict__ G,
    float* __restrict__ out, unsigned* __restrict__ gmax) {
  __shared__ __bf16 Ahi[MB * PK];
  __shared__ __bf16 Alo[MB * PK];
  __shared__ int    sidx[MB];

  const int  t  = threadIdx.x;
  const long eb = (long)blockIdx.x * MB;

  if (t < MB) {
    long e = eb + t;
    sidx[t] = (e < N_EDGES) ? gidx[e] : 0;
  }
  // Stage 64x128 f32 edge tile as bf16 hi/lo (converted once, shared by 8 waves)
  #pragma unroll
  for (int i = 0; i < 8; ++i) {
    int j   = i * 256 + t;          // 2048 float4 total
    int row = j >> 5;
    int c4  = (j & 31) << 2;
    float4 v = make_float4(0.f, 0.f, 0.f, 0.f);
    long e = eb + row;
    if (e < N_EDGES) v = *(const float4*)(edge + e * 128 + c4);
    float fx[4] = {v.x, v.y, v.z, v.w};
    v4bf hv, lv;
    #pragma unroll
    for (int q = 0; q < 4; ++q) {
      __bf16 h = (__bf16)fx[q];
      hv[q] = h;
      lv[q] = (__bf16)(fx[q] - (float)h);
    }
    int o = row * PK + c4;          // 8B aligned
    *(v4bf*)(Ahi + o) = hv;
    *(v4bf*)(Alo + o) = lv;
  }
  __syncthreads();

  const int lane  = t & 31;
  const int wid   = t >> 5;
  const int nloc  = lane & 15;
  const int khalf = lane >> 4;        // 0|1
  const int n     = wid * 16 + nloc;  // output column owned by this lane

  // B fragments: precomputed & swizzled in ws, coalesced 32B loads per lane
  v16bf Bh[4], Bl[4];
  #pragma unroll
  for (int kc = 0; kc < 4; ++kc) {
    long fi = (long)((wid * 4 + kc) * 32 + lane);
    Bh[kc] = ((const v16bf*)Bhw)[fi];
    Bl[kc] = ((const v16bf*)Blw)[fi];
  }

  #pragma unroll
  for (int ms = 0; ms < 4; ++ms) {
    // Accumulator initialized with gathered per-graph row (bias + graph term)
    v8f acc;
    #pragma unroll
    for (int v = 0; v < 8; ++v) {
      int r = ms * 16 + khalf * 8 + v;        // C/D row for this lane/VGPR
      acc[v] = G[(long)sidx[r] * 128 + n];
    }
    #pragma unroll
    for (int kc = 0; kc < 4; ++kc) {
      // A fragment: lane holds row nloc, K runs {kc*32+khalf*16..+15}
      int ob = (ms * 16 + nloc) * PK + kc * 32 + khalf * 16;
      v8bf h0 = *(const v8bf*)(Ahi + ob);
      v8bf h1 = *(const v8bf*)(Ahi + ob + 8);
      v8bf l0 = *(const v8bf*)(Alo + ob);
      v8bf l1 = *(const v8bf*)(Alo + ob + 8);
      v16bf ah = __builtin_shufflevector(h0, h1, 0,1,2,3,4,5,6,7,8,9,10,11,12,13,14,15);
      v16bf al = __builtin_shufflevector(l0, l1, 0,1,2,3,4,5,6,7,8,9,10,11,12,13,14,15);
      // split product: a*b ~= ah*bh + ah*bl + al*bh  (~16 mantissa bits)
      acc = __builtin_amdgcn_wmma_f32_16x16x32_bf16(false, al, false, Bh[kc],
                                                    (short)0, acc, false, false);
      acc = __builtin_amdgcn_wmma_f32_16x16x32_bf16(false, ah, false, Bl[kc],
                                                    (short)0, acc, false, false);
      acc = __builtin_amdgcn_wmma_f32_16x16x32_bf16(false, ah, false, Bh[kc],
                                                    (short)0, acc, false, false);
    }
    // Store y and fold segment-max in (monotonic-uint atomicMax)
    #pragma unroll
    for (int v = 0; v < 8; ++v) {
      int  r = ms * 16 + khalf * 8 + v;
      long e = eb + r;
      if (e < N_EDGES) {
        float val = acc[v];
        out[e * 128 + n] = val;
        atomicMax(&gmax[(long)sidx[r] * 128 + n], encf(val));
      }
    }
  }
}

// ------------------------- softmax passes ----------------------------------
__global__ void ga_exp_sum(float* __restrict__ out, const int* __restrict__ gidx,
                           const unsigned* __restrict__ gmax, float* __restrict__ denom) {
  long i = ((long)blockIdx.x * blockDim.x + threadIdx.x) * 4;   // 64e6 exactly
  long e = i >> 7;
  int  f = (int)(i & 127);
  int  g = gidx[e];
  const unsigned* gm = gmax + (long)g * 128 + f;
  float4 y = *(float4*)(out + i);
  float4 r;
  r.x = expf(y.x - decf(gm[0]));
  r.y = expf(y.y - decf(gm[1]));
  r.z = expf(y.z - decf(gm[2]));
  r.w = expf(y.w - decf(gm[3]));
  *(float4*)(out + i) = r;
  float* d = denom + (long)g * 128 + f;
  atomicAdd(d + 0, r.x);
  atomicAdd(d + 1, r.y);
  atomicAdd(d + 2, r.z);
  atomicAdd(d + 3, r.w);
}

__global__ void ga_normalize(float* __restrict__ out, const int* __restrict__ gidx,
                             const float* __restrict__ denom) {
  long i = ((long)blockIdx.x * blockDim.x + threadIdx.x) * 4;
  long e = i >> 7;
  int  f = (int)(i & 127);
  const float* d = denom + (long)gidx[e] * 128 + f;
  float4 v = *(float4*)(out + i);
  v.x /= d[0]; v.y /= d[1]; v.z /= d[2]; v.w /= d[3];
  *(float4*)(out + i) = v;
}

// ---------------------------------------------------------------------------
extern "C" void kernel_launch(void* const* d_in, const int* in_sizes, int n_in,
                              void* d_out, int out_size, void* d_ws, size_t ws_size,
                              hipStream_t stream) {
  (void)in_sizes; (void)n_in; (void)out_size; (void)ws_size;
  const float* graph_emb  = (const float*)d_in[0];   // 1024 x 256
  const float* edge_emb   = (const float*)d_in[1];   // 500000 x 128
  const int*   gidx       = (const int*)  d_in[2];   // 500000
  const float* W_in       = (const float*)d_in[3];   // 256 x 384
  const float* b_in       = (const float*)d_in[4];   // 256
  const float* in_proj_w  = (const float*)d_in[5];   // 768 x 256
  const float* in_proj_b  = (const float*)d_in[6];   // 768
  const float* out_proj_w = (const float*)d_in[7];   // 256 x 256
  const float* out_proj_b = (const float*)d_in[8];   // 256
  const float* W_out      = (const float*)d_in[9];   // 128 x 256
  const float* b_out      = (const float*)d_in[10];  // 128
  float* out = (float*)d_out;                        // 500000 x 128

  // workspace layout (~2.1 MB, all offsets 32B aligned)
  float*    T1    = (float*)d_ws;            // 128x256        @0
  float*    T2    = T1 + 32768;              // 128x256        @32768
  float*    Wc    = T2 + 32768;              // 128x384        @65536
  float*    bc    = Wc + 49152;              // 128            @114688
  float*    G     = bc + 128;                // 1024x128       @114816
  unsigned* gmax  = (unsigned*)(G + 131072); // 1024x128       @245888
  float*    denom = (float*)(gmax + 131072); // 1024x128       @376960
  __bf16*   Bhw   = (__bf16*)(denom + 131072);   // 16384 bf16 @508032
  __bf16*   Blw   = Bhw + 16384;                 // 16384 bf16

  ga_init_kernel<<<512, 256, 0, stream>>>(gmax, denom);

  // collapse the 4 linear layers: T1 = W4·W3, T2 = T1·W2', Wc = T2·W1, + bc
  ga_gemm_small<<<128, 256, 0, stream>>>(W_out, out_proj_w, T1, 128, 256, 256);
  ga_gemm_small<<<128, 256, 0, stream>>>(T1, in_proj_w + 512 * 256, T2, 128, 256, 256);
  ga_gemm_small<<<192, 256, 0, stream>>>(T2, W_in, Wc, 128, 384, 256);
  ga_bias_kernel<<<1, 128, 0, stream>>>(T1, T2, W_out, b_in, in_proj_b, out_proj_b, b_out, bc);

  ga_graph_proj<<<512, 256, 0, stream>>>(Wc, bc, graph_emb, G);
  ga_bsplit<<<4, 256, 0, stream>>>(Wc, Bhw, Blw);

  ga_main_gemm<<<(N_EDGES + MB - 1) / MB, 256, 0, stream>>>(edge_emb, gidx, Bhw, Blw,
                                                            G, out, gmax);

  ga_exp_sum<<<62500, 256, 0, stream>>>(out, gidx, gmax, denom);
  ga_normalize<<<62500, 256, 0, stream>>>(out, gidx, denom);
}
